// QueryInit_1005022347952
// MI455X (gfx1250) — compile-verified, matrix operand order checked by reading, not run on hardware
//
#include <hip/hip_runtime.h>
#include <hip/hip_bf16.h>

typedef __attribute__((ext_vector_type(16))) _Float16 v16h;
typedef __attribute__((ext_vector_type(8)))  float    v8f;
typedef unsigned int u32;

#define NPTS 120000
#define CCH  32
#define DZ_  41
#define DY_  400
#define DX_  352
#define YX_  (DY_*DX_)            /* 140800  */
#define VOX_ (DZ_*YX_)            /* 5772800 */
#define NF_  60000                /* N*0.5   */
#define MROW (NF_*27)             /* 1620000 */
#define ROWS (MROW + (NPTS-NF_))  /* 1680000 */
#define NTILE (NPTS/16)           /* 7500    */

#define RS_CHUNK 1024
#define RS_NCH   ((NPTS + RS_CHUNK - 1)/RS_CHUNK)   /* 118  */
#define SC_CHUNK 2048
#define SC_NCH   ((VOX_ + SC_CHUNK - 1)/SC_CHUNK)   /* 2819 */

// ---------------------------------------------------------------- utilities
__global__ void k_zero4(float4* p, size_t n4){
  size_t i  = (size_t)blockIdx.x*blockDim.x + threadIdx.x;
  size_t st = (size_t)gridDim.x*blockDim.x;
  float4 z = make_float4(0.f,0.f,0.f,0.f);
  for (; i < n4; i += st) p[i] = z;
}

__global__ void k_init(int* vox, u32* flags, u32* isFore){
  size_t i  = (size_t)blockIdx.x*blockDim.x + threadIdx.x;
  size_t st = (size_t)gridDim.x*blockDim.x;
  for (; i < (size_t)VOX_; i += st){
    vox[i]   = -1;
    flags[i] = 0u;
    if (i < (size_t)NPTS) isFore[i] = 0u;
  }
}

__global__ void k_scatter(const int* __restrict__ xi, int* __restrict__ vox){
  int p = blockIdx.x*blockDim.x + threadIdx.x;
  if (p < NPTS){
    int z = xi[p*4+1], y = xi[p*4+2], x = xi[p*4+3];
    vox[z*YX_ + y*DX_ + x] = p;
  }
}

// Pre-arrange W (27,32,27) into WMMA B fragments: [k][ntile][lane] -> v16h.
// Lane l holds column n = ntile*16 + (l&15); K pattern: lanes<16 -> {0..7,16..23},
// lanes>=16 -> {8..15,24..31}  (mirrors the 16-bit A-matrix layout).
__global__ void k_arrange(const float* __restrict__ w, v16h* __restrict__ wfrag){
  int e = blockIdx.x*blockDim.x + threadIdx.x;
  if (e < 27*2*32){
    int k = e >> 6, rem = e & 63, tile = rem >> 5, lane = rem & 31;
    int n  = tile*16 + (lane & 15);
    int hi = (lane >> 4) & 1;
    int c0 = hi ? 8 : 0, c1 = hi ? 24 : 16;
    v16h f = {};
    if (n < 27){
      #pragma unroll
      for (int i = 0; i < 8; ++i){
        f[i]   = (_Float16)w[((size_t)(k*32 + c0 + i))*27 + n];
        f[8+i] = (_Float16)w[((size_t)(k*32 + c1 + i))*27 + n];
      }
    }
    wfrag[e] = f;
  }
}

// ------------------------------------------------- gather + WMMA GEMM stage
// One wave per 16-point tile. imps = sum_k gather_k(16x32 f16) @ W_k(32x32 f16),
// accumulated in f32 across 27 offsets via v_wmma_f32_16x16x32_f16.
__global__ void __launch_bounds__(256) k_gemm(
    const float* __restrict__ xf, const int* __restrict__ xi,
    const int* __restrict__ vox, const v16h* __restrict__ wfrag,
    float* __restrict__ mv, u32* __restrict__ kb, float* __restrict__ feats)
{
  __shared__ float simp[8][16][33];   // per-wave imps tile scratch
  const int w    = threadIdx.x >> 5;
  const int lane = threadIdx.x & 31;
  const int r    = lane & 15;
  const int hi   = lane >> 4;
  const int wavesTotal = gridDim.x * 8;

  for (int t = blockIdx.x*8 + w; t < NTILE; t += wavesTotal){
    const int p  = t*16 + r;
    const int cz = xi[p*4+1], cy = xi[p*4+2], cx = xi[p*4+3];
    v8f acc0 = {}, acc1 = {};

    #pragma unroll 1
    for (int k = 0; k < 27; ++k){
      const int dz = k/9 - 1, dy = (k/3)%3 - 1, dx = k%3 - 1;
      const int nz = cz+dz, ny = cy+dy, nx = cx+dx;
      int nid = -1;
      if ((u32)nz < (u32)DZ_ && (u32)ny < (u32)DY_ && (u32)nx < (u32)DX_)
        nid = vox[nz*YX_ + ny*DX_ + nx];

      v16h a = {};
      if (nid >= 0){
        const float4* fr = (const float4*)(xf + (size_t)nid*CCH);
        const int b0 = hi ? 2 : 0;   // channels {0..7} or {8..15}
        const int b1 = hi ? 6 : 4;   // channels {16..23} or {24..31}
        float4 f0 = fr[b0], f1 = fr[b0+1], f2 = fr[b1], f3 = fr[b1+1];
        a[0]=(_Float16)f0.x;  a[1]=(_Float16)f0.y;  a[2]=(_Float16)f0.z;  a[3]=(_Float16)f0.w;
        a[4]=(_Float16)f1.x;  a[5]=(_Float16)f1.y;  a[6]=(_Float16)f1.z;  a[7]=(_Float16)f1.w;
        a[8]=(_Float16)f2.x;  a[9]=(_Float16)f2.y;  a[10]=(_Float16)f2.z; a[11]=(_Float16)f2.w;
        a[12]=(_Float16)f3.x; a[13]=(_Float16)f3.y; a[14]=(_Float16)f3.z; a[15]=(_Float16)f3.w;
      }
      v16h b0v = wfrag[(k*2+0)*32 + lane];
      v16h b1v = wfrag[(k*2+1)*32 + lane];
      acc0 = __builtin_amdgcn_wmma_f32_16x16x32_f16(false, a, false, b0v, (short)0, acc0, false, false);
      acc1 = __builtin_amdgcn_wmma_f32_16x16x32_f16(false, a, false, b1v, (short)0, acc1, false, false);
    }

    // C/D layout: VGPR j, lanes 0-15 -> (M=j, N=lane); lanes 16-31 -> (M=j+8, N=lane-16)
    const int mbase = 8*hi;
    #pragma unroll
    for (int j = 0; j < 8; ++j){
      simp[w][mbase+j][r]    = acc0[j];
      simp[w][mbase+j][16+r] = acc1[j];
    }
    asm volatile("s_wait_dscnt 0x0" ::: "memory");

    const float iv = simp[w][r][26];          // column 26 == OFFS[26] logit
    const float s  = 1.0f / (1.0f + __expf(-iv));
    if (hi == 0){
      mv[p] = s;
      u32 bits = 0;
      #pragma unroll 1
      for (int j = 0; j < 26; ++j)
        if (simp[w][r][j] >= 0.0f) bits |= (1u << j);  // sigmoid>=0.5 <=> logit>=0
      kb[p] = bits;
    }
    const float4* src = (const float4*)(xf + (size_t)p*CCH) + hi*4;
    float4*       dst = (float4*)(feats + (size_t)p*CCH) + hi*4;
    #pragma unroll
    for (int q = 0; q < 4; ++q){
      float4 v = src[q];
      v.x *= s; v.y *= s; v.z *= s; v.w *= s;
      dst[q] = v;
    }
    asm volatile("s_wait_dscnt 0x0" ::: "memory");  // simp reused next tile
  }
}

// -------------------------------------------- stable LSD radix sort (desc mv)
__global__ void k_keys(const float* __restrict__ mv, u32* __restrict__ key, u32* __restrict__ idx){
  int p = blockIdx.x*blockDim.x + threadIdx.x;
  if (p < NPTS){
    u32 u   = __float_as_uint(mv[p]);
    u32 ord = u ^ ((u >> 31) ? 0xFFFFFFFFu : 0x80000000u); // ascending float order
    key[p] = ~ord;                                          // ascending sort -> descending mv
    idx[p] = (u32)p;
  }
}

__global__ void k_hist(const u32* __restrict__ key, u32* __restrict__ gh, int n, int shift){
  __shared__ u32 cnt[16];
  if (threadIdx.x < 16) cnt[threadIdx.x] = 0;
  __syncthreads();
  int base = blockIdx.x*RS_CHUNK + threadIdx.x*4;
  for (int i = 0; i < 4; ++i){
    int p = base + i;
    if (p < n) atomicAdd(&cnt[(key[p] >> shift) & 15], 1u);
  }
  __syncthreads();
  if (threadIdx.x < 16) gh[threadIdx.x*RS_NCH + blockIdx.x] = cnt[threadIdx.x];
}

__global__ void k_scan16(u32* gh){
  __shared__ u32 tot[16], base[16];
  int b = threadIdx.x;
  if (b < 16){
    u32 run = 0;
    for (int i = 0; i < RS_NCH; ++i){ u32 v = gh[b*RS_NCH+i]; gh[b*RS_NCH+i] = run; run += v; }
    tot[b] = run;
  }
  __syncthreads();
  if (b == 0){ u32 run = 0; for (int i = 0; i < 16; ++i){ base[i] = run; run += tot[i]; } }
  __syncthreads();
  if (b < 16) for (int i = 0; i < RS_NCH; ++i) gh[b*RS_NCH+i] += base[b];
}

__global__ void k_scat(const u32* __restrict__ ki, const u32* __restrict__ ii,
                       u32* __restrict__ ko, u32* __restrict__ io,
                       const u32* __restrict__ gh, int n, int shift){
  __shared__ u32 thc[16][256];
  int t = threadIdx.x;
  u32 k[4]; u32 d[4]; u32 lc[16];
  for (int b = 0; b < 16; ++b) lc[b] = 0;
  int base = blockIdx.x*RS_CHUNK + t*4;
  for (int i = 0; i < 4; ++i){
    int p = base + i;
    if (p < n){ k[i] = ki[p]; d[i] = (k[i] >> shift) & 15; lc[d[i]]++; }
    else d[i] = 16;
  }
  for (int b = 0; b < 16; ++b) thc[b][t] = lc[b];
  __syncthreads();
  if (t < 16){ u32 run = 0; for (int i = 0; i < 256; ++i){ u32 v = thc[t][i]; thc[t][i] = run; run += v; } }
  __syncthreads();
  u32 off[16];
  for (int b = 0; b < 16; ++b) off[b] = gh[b*RS_NCH + blockIdx.x] + thc[b][t];
  for (int i = 0; i < 4; ++i){
    int p = base + i;
    if (p < n){ u32 b = d[i]; u32 pos = off[b]++; ko[pos] = k[i]; io[pos] = ii[p]; }
  }
}

// fore marking + back-half output rows
__global__ void k_post(const u32* __restrict__ order, const float* __restrict__ feats,
                       const int* __restrict__ xi, u32* __restrict__ isFore,
                       float* __restrict__ outF, int* __restrict__ outI, float* __restrict__ outV){
  int i = blockIdx.x*blockDim.x + threadIdx.x;
  if (i < NPTS){
    int pid = (int)order[i];
    if (i < NF_) isFore[pid] = 1u;
    else {
      int row = MROW + (i - NF_);
      const float4* s4 = (const float4*)(feats + (size_t)pid*CCH);
      float4*       d4 = (float4*)(outF + (size_t)row*CCH);
      #pragma unroll
      for (int q = 0; q < 8; ++q) d4[q] = s4[q];
      outI[(size_t)row*4+0] = 0;
      outI[(size_t)row*4+1] = xi[pid*4+1];
      outI[(size_t)row*4+2] = xi[pid*4+2];
      outI[(size_t)row*4+3] = xi[pid*4+3];
      outV[row] = 1.0f;
    }
  }
}

// mark valid voxels: fore own voxels (j==26 slot) + masked in-bounds candidates
__global__ void k_mark(const u32* __restrict__ order, const int* __restrict__ xi,
                       const u32* __restrict__ kb, u32* __restrict__ flags){
  int t = blockIdx.x*blockDim.x + threadIdx.x;
  if (t < MROW){
    int i = t/27, j = t%27;
    int pid = (int)order[i];
    int z = xi[pid*4+1], y = xi[pid*4+2], x = xi[pid*4+3];
    if (j == 26){
      flags[z*YX_ + y*DX_ + x] = 1u;
    } else if ((kb[pid] >> j) & 1u){
      int oi = (j < 13) ? j : j + 1;               // KOFFS[j] = OFFS without center
      int nz = z + oi/9 - 1, ny = y + (oi/3)%3 - 1, nx = x + oi%3 - 1;
      if (nz > 0 && ny > 0 && nx > 0 && nz < DZ_ && ny < DY_ && nx < DX_)
        flags[nz*YX_ + ny*DX_ + nx] = 1u;
    }
  }
}

// exclusive prefix sum over VOX_ flags -> segment ranks (ascending linear-key order)
__global__ void k_ssum(const u32* __restrict__ flags, u32* __restrict__ part){
  __shared__ u32 s[256];
  int t = threadIdx.x;
  size_t base = (size_t)blockIdx.x*SC_CHUNK + (size_t)t*8;
  u32 sum = 0;
  for (int i = 0; i < 8; ++i){ size_t p = base + i; if (p < (size_t)VOX_) sum += flags[p]; }
  s[t] = sum; __syncthreads();
  for (int st = 128; st > 0; st >>= 1){ if (t < st) s[t] += s[t+st]; __syncthreads(); }
  if (t == 0) part[blockIdx.x] = s[0];
}

__global__ void k_spart(u32* part){
  u32 run = 0;
  for (int i = 0; i < SC_NCH; ++i){ u32 v = part[i]; part[i] = run; run += v; }
}

__global__ void k_swrite(const u32* __restrict__ flags, const u32* __restrict__ part,
                         u32* __restrict__ ranks){
  __shared__ u32 s[256];
  int t = threadIdx.x;
  size_t base = (size_t)blockIdx.x*SC_CHUNK + (size_t)t*8;
  u32 f[8]; u32 sum = 0;
  for (int i = 0; i < 8; ++i){ size_t p = base + i; f[i] = (p < (size_t)VOX_) ? flags[p] : 0u; sum += f[i]; }
  s[t] = sum; __syncthreads();
  for (int st = 1; st < 256; st <<= 1){
    u32 v = (t >= st) ? s[t-st] : 0u;
    __syncthreads();
    s[t] += v;
    __syncthreads();
  }
  u32 run = part[blockIdx.x] + (s[t] - sum);
  for (int i = 0; i < 8; ++i){
    size_t p = base + i;
    if (p < (size_t)VOX_){ ranks[p] = run; run += f[i]; }
  }
}

// emit deduped segments: coords from voxel id; feats only if voxel hosts a fore point
__global__ void k_dedup(const u32* __restrict__ flags, const u32* __restrict__ ranks,
                        const int* __restrict__ vox, const u32* __restrict__ isFore,
                        const float* __restrict__ feats,
                        float* __restrict__ outF, int* __restrict__ outI, float* __restrict__ outV){
  size_t v  = (size_t)blockIdx.x*blockDim.x + threadIdx.x;
  size_t st = (size_t)gridDim.x*blockDim.x;
  for (; v < (size_t)VOX_; v += st){
    if (!flags[v]) continue;
    u32 s = ranks[v];
    int z = (int)(v / YX_); int rem = (int)(v % YX_); int y = rem / DX_; int x = rem % DX_;
    outI[(size_t)s*4+0] = 0; outI[(size_t)s*4+1] = z;
    outI[(size_t)s*4+2] = y; outI[(size_t)s*4+3] = x;
    outV[s] = 1.0f;
    int pid = vox[v];
    if (pid >= 0 && isFore[pid]){
      const float4* s4 = (const float4*)(feats + (size_t)pid*CCH);
      float4*       d4 = (float4*)(outF + (size_t)s*CCH);
      #pragma unroll
      for (int q = 0; q < 8; ++q) d4[q] = s4[q];
    }
  }
}

// ------------------------------------------------------------------- driver
extern "C" void kernel_launch(void* const* d_in, const int* in_sizes, int n_in,
                              void* d_out, int out_size, void* d_ws, size_t ws_size,
                              hipStream_t stream)
{
  (void)in_sizes; (void)n_in; (void)out_size; (void)ws_size;
  const float* xf = (const float*)d_in[0];   // x_features (N,32)
  const int*   xi = (const int*)d_in[1];     // x_indices  (N,4)
  const float* wi = (const float*)d_in[2];   // w_imp      (27,32,27)

  char* ws = (char*)d_ws;
  size_t off = 0;
  auto take = [&](size_t bytes)->char*{
    char* p = ws + off;
    off = (off + bytes + 255) & ~(size_t)255;
    return p;
  };
  int*   vox    = (int*)  take((size_t)VOX_*4);
  u32*   flags  = (u32*)  take((size_t)VOX_*4);
  u32*   ranks  = (u32*)  take((size_t)VOX_*4);
  float* mv     = (float*)take((size_t)NPTS*4);
  u32*   kb     = (u32*)  take((size_t)NPTS*4);
  u32*   isFore = (u32*)  take((size_t)NPTS*4);
  float* feats  = (float*)take((size_t)NPTS*CCH*4);
  u32*   keyA   = (u32*)  take((size_t)NPTS*4);
  u32*   idxA   = (u32*)  take((size_t)NPTS*4);
  u32*   keyB   = (u32*)  take((size_t)NPTS*4);
  u32*   idxB   = (u32*)  take((size_t)NPTS*4);
  u32*   gh     = (u32*)  take((size_t)16*RS_NCH*4);
  u32*   part   = (u32*)  take((size_t)SC_NCH*4);
  v16h*  wfrag  = (v16h*) take((size_t)27*2*32*32);

  float* outF = (float*)d_out;
  int*   outI = (int*)(outF + (size_t)ROWS*32);
  float* outV = outF + (size_t)ROWS*36;

  size_t totF4 = (size_t)ROWS*37/4;   // 15,540,000
  k_zero4  <<<4096, 256, 0, stream>>>((float4*)d_out, totF4);
  k_init   <<<4096, 256, 0, stream>>>(vox, flags, isFore);
  k_scatter<<<(NPTS+255)/256, 256, 0, stream>>>(xi, vox);
  k_arrange<<<7, 256, 0, stream>>>(wi, wfrag);
  k_gemm   <<<938, 256, 0, stream>>>(xf, xi, vox, wfrag, mv, kb, feats);
  k_keys   <<<(NPTS+255)/256, 256, 0, stream>>>(mv, keyA, idxA);

  u32 *ki = keyA, *ii = idxA, *ko = keyB, *io = idxB;
  for (int pass = 0; pass < 8; ++pass){
    int shift = pass*4;
    k_hist  <<<RS_NCH, 256, 0, stream>>>(ki, gh, NPTS, shift);
    k_scan16<<<1, 256, 0, stream>>>(gh);
    k_scat  <<<RS_NCH, 256, 0, stream>>>(ki, ii, ko, io, gh, NPTS, shift);
    u32* t;
    t = ki; ki = ko; ko = t;
    t = ii; ii = io; io = t;
  }

  k_post  <<<(NPTS+255)/256, 256, 0, stream>>>(ii, feats, xi, isFore, outF, outI, outV);
  k_mark  <<<(MROW+255)/256, 256, 0, stream>>>(ii, xi, kb, flags);
  k_ssum  <<<SC_NCH, 256, 0, stream>>>(flags, part);
  k_spart <<<1, 1, 0, stream>>>(part);
  k_swrite<<<SC_NCH, 256, 0, stream>>>(flags, part, ranks);
  k_dedup <<<4096, 256, 0, stream>>>(flags, ranks, vox, isFore, feats, outF, outI, outV);
}
